// MultiHeadAttention_24240795419256
// MI455X (gfx1250) — compile-verified, matrix operand order checked by reading, not run on hardware
//
#include <hip/hip_runtime.h>
#include <hip/hip_bf16.h>
#include <stdint.h>

// MultiHeadAttention forward for MI455X (gfx1250), wave32 + WMMA f16 path.
// Pipeline: convert -> QKV GEMM (async-to-LDS A staging) ->
//           fused softmax-attention (materializes attn) -> proj+residual+LN.

#define NHEADC 16
#define NIN    1024
#define DKC    64
#define BATCH  4
#define SEQ    2048
#define MTOT   (BATCH*SEQ)   // 8192

typedef __attribute__((ext_vector_type(16))) _Float16 v16h;
typedef __attribute__((ext_vector_type(8)))  float    v8f;

// --- WMMA helpers ---------------------------------------------------------

static __device__ __forceinline__ v8f wmma16(v16h a, v16h b, v8f c) {
  // D(16x16 f32) = A(16x32 f16) * B(32x16 f16) + C
  return __builtin_amdgcn_wmma_f32_16x16x32_f16(
      /*neg_a=*/false, a, /*neg_b=*/false, b,
      /*c_mod=*/(short)0, c, /*reuse_a=*/false, /*reuse_b=*/false);
}

// K-offset of dword-pair d (0..7) for half-group h (lane>>4), per CDNA5 ISA
// 16-bit A-matrix layout: VGPR0..3 -> K {0,2,4,6}+8h ; VGPR4..7 -> K {16,18,20,22}+8h
static __device__ __forceinline__ int koff_pair(int d, int h) {
  return ((d < 4) ? (2 * d) : (16 + 2 * (d - 4))) + 8 * h;
}

// Load a 16x32 (f16) fragment. Works for A (row = M = lane&15) and for B when
// B is stored K-major (row = N = lane&15, elements = K). `ld` = row stride.
// Compiler vectorizes this into two b128 loads per lane (global or ds).
static __device__ __forceinline__ v16h load_frag(const _Float16* base, int ld, int lane) {
  const int r = lane & 15, h = lane >> 4;
  const _Float16* p = base + (size_t)r * ld;
  v16h f;
#pragma unroll
  for (int d = 0; d < 8; ++d) {
    const int k = koff_pair(d, h);
    f[2 * d]     = p[k];
    f[2 * d + 1] = p[k + 1];
  }
  return f;
}

// --- gfx1250 async copy: global -> LDS (ASYNCcnt path) --------------------

static __device__ __forceinline__ void async_g2lds_b128(const void* gptr, void* ldsptr) {
  // GLOBAL_LOAD_ASYNC_TO_LDS_B128, GV mode: VDST = per-lane LDS byte address,
  // VADDR = 64-bit global address. Tracked by ASYNCcnt.
  uint32_t l = (uint32_t)(uintptr_t)ldsptr;
  uint64_t g = (uint64_t)(uintptr_t)gptr;
  asm volatile("global_load_async_to_lds_b128 %0, %1, off"
               :: "v"(l), "v"(g) : "memory");
}
static __device__ __forceinline__ void wait_async0() {
  asm volatile("s_wait_asynccnt 0x0" ::: "memory");
}

// --- cross-lane reductions (wave32) ---------------------------------------

static __device__ __forceinline__ float red16_max(float v) {
#pragma unroll
  for (int m = 1; m < 16; m <<= 1) v = fmaxf(v, __shfl_xor(v, m, 32));
  return v;
}
static __device__ __forceinline__ float red16_sum(float v) {
#pragma unroll
  for (int m = 1; m < 16; m <<= 1) v += __shfl_xor(v, m, 32);
  return v;
}
static __device__ __forceinline__ float red32_sum(float v) {
#pragma unroll
  for (int m = 1; m < 32; m <<= 1) v += __shfl_xor(v, m, 32);
  return v;
}

// --- stage 1: conversions -------------------------------------------------

__global__ void cvt_f32_to_f16(const float* __restrict__ src,
                               _Float16* __restrict__ dst, int n) {
  int i = blockIdx.x * blockDim.x + threadIdx.x;
  if (i < n) dst[i] = (_Float16)src[i];
}

// src: [NIN (in), NIN (out)] f32 row-major -> dst: [out, in] f16 (K-major B)
__global__ void transpose_cvt(const float* __restrict__ src,
                              _Float16* __restrict__ dst) {
  int idx = blockIdx.x * blockDim.x + threadIdx.x;   // over NIN*NIN
  int i = idx >> 10;        // input row (K)
  int o = idx & 1023;       // output col (N)
  dst[(size_t)o * NIN + i] = (_Float16)src[idx];
}

// --- stage 2: QKV projections --------------------------------------------
// One wave = one 16x16 C tile. grid = (MTOT/16, NIN/64, 3), block = 128 (4 waves).
// The 16x1024 A panel is staged to LDS once per block via async loads.

__global__ void qkv_gemm(const _Float16* __restrict__ xh,
                         const _Float16* __restrict__ wq_t,
                         const _Float16* __restrict__ wk_t,
                         const _Float16* __restrict__ wv_t,
                         _Float16* __restrict__ q,
                         _Float16* __restrict__ k,
                         _Float16* __restrict__ vt) {
  __shared__ __align__(16) _Float16 aTile[16 * NIN];   // 32 KB
  const int lane = threadIdx.x & 31;
  const int wave = threadIdx.x >> 5;                 // 0..3
  const int m0 = blockIdx.x * 16;
  const int n0 = (blockIdx.y * 4 + wave) * 16;
  const int which = blockIdx.z;                      // 0=Q 1=K 2=V
  const _Float16* wt = (which == 0) ? wq_t : (which == 1) ? wk_t : wv_t;

  // cooperative async stage of A panel: 2048 x 16B chunks, 128 threads
  {
    const _Float16* src = xh + (size_t)m0 * NIN;     // contiguous 32 KB
#pragma unroll
    for (int c = threadIdx.x; c < 2048; c += 128) {
      async_g2lds_b128(src + (size_t)c * 8, aTile + (size_t)c * 8);
    }
    wait_async0();
    __syncthreads();
  }

  const _Float16* bbase = wt + (size_t)n0 * NIN;
  v8f acc = {};
  for (int kk = 0; kk < NIN; kk += 32) {
    __builtin_prefetch(bbase + kk + 512, 0, 0);      // global_prefetch_b8
    v16h a = load_frag(aTile + kk, NIN, lane);       // ds path
    v16h b = load_frag(bbase + kk, NIN, lane);       // vmem path
    acc = wmma16(a, b, acc);
  }

  const float scale = (which == 0) ? 0.125f : 1.0f;  // fold 1/sqrt(64) into Q
  const int h = lane >> 4;
  const int col = n0 + (lane & 15);
  const int head = col >> 6, d = col & 63;
#pragma unroll
  for (int r = 0; r < 8; ++r) {
    int m = m0 + r + 8 * h;
    int bb = m >> 11, s = m & (SEQ - 1);
    size_t bh = (size_t)(bb * NHEADC + head);
    _Float16 val = (_Float16)(acc[r] * scale);
    if (which == 2)      vt[(bh * DKC + d) * SEQ + s] = val;     // V transposed
    else if (which == 0) q[(bh * SEQ + s) * DKC + d] = val;
    else                 k[(bh * SEQ + s) * DKC + d] = val;
  }
}

// --- stage 3: fused attention (scores -> softmax -> attn out -> P*V) ------
// grid = (B*H, SEQ/16), block = 256 (8 waves). Wave w owns columns [w*256, w*256+256).

__global__ void attn_kernel(const _Float16* __restrict__ q,
                            const _Float16* __restrict__ kmat,
                            const _Float16* __restrict__ vt,
                            float* __restrict__ attn_out,
                            _Float16* __restrict__ yh) {
  __shared__ _Float16 pBuf[16 * SEQ];     // 64 KB: f16 probabilities, 16 rows x SEQ
  __shared__ float sRed[2][8 * 16];       // cross-wave max / sum

  const int lane = threadIdx.x & 31;
  const int wave = threadIdx.x >> 5;      // 0..7
  const int h    = lane >> 4;
  const int ln   = lane & 15;
  const int bh   = blockIdx.x;            // 0..63
  const int s0   = blockIdx.y * 16;       // query-row block

  // ---- phase 1: scores c[t] = (Q/temp) * K^T for 16 tiles of 16 cols -----
  const _Float16* qbase = q + ((size_t)bh * SEQ + s0) * DKC;
  v16h a0 = load_frag(qbase,      DKC, lane);
  v16h a1 = load_frag(qbase + 32, DKC, lane);
  v8f c[16];
#pragma unroll
  for (int t = 0; t < 16; ++t) {
    int n0 = wave * 256 + t * 16;
    const _Float16* kb = kmat + ((size_t)bh * SEQ + n0) * DKC;
    v16h b0 = load_frag(kb,      DKC, lane);
    v16h b1 = load_frag(kb + 32, DKC, lane);
    v8f cc = {};
    cc = wmma16(a0, b0, cc);
    cc = wmma16(a1, b1, cc);
    c[t] = cc;
  }

  // ---- phase 2: softmax over full row (2048) -----------------------------
  float rmax[8];
#pragma unroll
  for (int r = 0; r < 8; ++r) {
    float m = c[0][r];
#pragma unroll
    for (int t = 1; t < 16; ++t) m = fmaxf(m, c[t][r]);
    rmax[r] = red16_max(m);               // max over this wave's 256 cols
  }
  if (ln == 0) {
#pragma unroll
    for (int r = 0; r < 8; ++r) sRed[0][wave * 16 + r + 8 * h] = rmax[r];
  }
  __syncthreads();
#pragma unroll
  for (int r = 0; r < 8; ++r) {
    float m = sRed[0][0 * 16 + r + 8 * h];
#pragma unroll
    for (int w2 = 1; w2 < 8; ++w2) m = fmaxf(m, sRed[0][w2 * 16 + r + 8 * h]);
    rmax[r] = m;                          // global row max
  }

  float rsum[8];
#pragma unroll
  for (int r = 0; r < 8; ++r) {
    float s = 0.f;
#pragma unroll
    for (int t = 0; t < 16; ++t) {
      float p = __expf(c[t][r] - rmax[r]);
      c[t][r] = p;
      s += p;
    }
    rsum[r] = red16_sum(s);
  }
  if (ln == 0) {
#pragma unroll
    for (int r = 0; r < 8; ++r) sRed[1][wave * 16 + r + 8 * h] = rsum[r];
  }
  __syncthreads();
#pragma unroll
  for (int r = 0; r < 8; ++r) {
    float s = 0.f;
#pragma unroll
    for (int w2 = 0; w2 < 8; ++w2) s += sRed[1][w2 * 16 + r + 8 * h];
    rsum[r] = 1.0f / s;
  }

  // normalize; write fp32 attn to d_out, f16 probs to LDS for the PV GEMM
  float* attnBase = attn_out + ((size_t)bh * SEQ + s0) * SEQ;
#pragma unroll
  for (int t = 0; t < 16; ++t) {
    int col = wave * 256 + t * 16 + ln;
#pragma unroll
    for (int r = 0; r < 8; ++r) {
      int row = r + 8 * h;
      float p = c[t][r] * rsum[r];
      attnBase[(size_t)row * SEQ + col] = p;
      pBuf[row * SEQ + col] = (_Float16)p;
    }
  }
  __syncthreads();

  // ---- phase 3: y_partial = P[16 x 256-chunk] * V^T ----------------------
  v8f y[4] = {};
  const int ksBase = wave * 256;
  for (int ks = 0; ks < 256; ks += 32) {
    v16h a = load_frag((const _Float16*)pBuf + ksBase + ks, SEQ, lane); // A from LDS
#pragma unroll
    for (int n = 0; n < 4; ++n) {
      const _Float16* vb = vt + ((size_t)bh * DKC + n * 16) * SEQ + ksBase + ks;
      v16h b = load_frag(vb, SEQ, lane);
      y[n] = wmma16(a, b, y[n]);
    }
  }

  __syncthreads();                 // all waves done reading pBuf
  float* sY = (float*)pBuf;        // alias first 32 KB for partial reduction
#pragma unroll
  for (int n = 0; n < 4; ++n) {
#pragma unroll
    for (int r = 0; r < 8; ++r) {
      sY[wave * 1024 + (r + 8 * h) * 64 + n * 16 + ln] = y[n][r];
    }
  }
  __syncthreads();

  // reduce 8 wave-partials, store f16 y block in [b, s, head*64+d] layout
  const int b_ = bh >> 4, head = bh & 15;
#pragma unroll
  for (int j = 0; j < 4; ++j) {
    int o = threadIdx.x * 4 + j;   // 0..1023 over 16x64 outputs
    float s = 0.f;
#pragma unroll
    for (int w2 = 0; w2 < 8; ++w2) s += sY[w2 * 1024 + o];
    int row = o >> 6, d = o & 63;
    int m = b_ * SEQ + s0 + row;
    yh[(size_t)m * NIN + head * 64 + d] = (_Float16)s;
  }
}

// --- stage 4: out projection + residual + LayerNorm -----------------------
// grid = (MTOT/16), block = 256 (8 waves); wave w owns output cols [w*128, w*128+128).
// A panel async-staged into LDS (aliases first half of sOut).

__global__ void proj_ln(const _Float16* __restrict__ yh,
                        const _Float16* __restrict__ wo_t,
                        const float* __restrict__ x,
                        const float* __restrict__ g,
                        const float* __restrict__ bta,
                        float* __restrict__ out) {
  __shared__ __align__(16) float sOut[16 * NIN];     // 64 KB
  _Float16* aTile = (_Float16*)sOut;                 // first 32 KB during GEMM
  const int lane = threadIdx.x & 31;
  const int wave = threadIdx.x >> 5;
  const int h = lane >> 4, ln = lane & 15;
  const int m0 = blockIdx.x * 16;
  const int n0 = wave * 128;

  // cooperative async stage of A panel (16 x 1024 f16 = 32 KB, contiguous)
  {
    const _Float16* src = yh + (size_t)m0 * NIN;
#pragma unroll
    for (int c = threadIdx.x; c < 2048; c += 256) {
      async_g2lds_b128(src + (size_t)c * 8, aTile + (size_t)c * 8);
    }
    wait_async0();
    __syncthreads();
  }

  v8f acc[8] = {};
  for (int kk = 0; kk < NIN; kk += 32) {
    __builtin_prefetch(wo_t + (size_t)n0 * NIN + kk + 512, 0, 0);
    v16h a = load_frag(aTile + kk, NIN, lane);       // ds path
#pragma unroll
    for (int t = 0; t < 8; ++t) {
      v16h b = load_frag(wo_t + (size_t)(n0 + t * 16) * NIN + kk, NIN, lane);
      acc[t] = wmma16(a, b, acc[t]);
    }
  }
  __syncthreads();               // all A reads done before overwriting sOut
#pragma unroll
  for (int t = 0; t < 8; ++t) {
#pragma unroll
    for (int r = 0; r < 8; ++r) {
      sOut[(r + 8 * h) * NIN + n0 + t * 16 + ln] = acc[t][r];
    }
  }
  __syncthreads();

  // LayerNorm: wave w handles rows {w, w+8}; residual added from x (f32)
  for (int rr = wave; rr < 16; rr += 8) {
    const int m = m0 + rr;
    float s1 = 0.f, s2 = 0.f;
    for (int j = 0; j < 32; ++j) {
      int cidx = lane + j * 32;
      float v = sOut[rr * NIN + cidx] + x[(size_t)m * NIN + cidx];
      s1 += v; s2 += v * v;
    }
    s1 = red32_sum(s1);
    s2 = red32_sum(s2);
    float mean = s1 * (1.0f / NIN);
    float var  = s2 * (1.0f / NIN) - mean * mean;
    float inv  = rsqrtf(var + 1e-6f);
    for (int j = 0; j < 32; ++j) {
      int cidx = lane + j * 32;
      float v = sOut[rr * NIN + cidx] + x[(size_t)m * NIN + cidx];
      out[(size_t)m * NIN + cidx] = (v - mean) * inv * g[cidx] + bta[cidx];
    }
  }
}

// --- launcher -------------------------------------------------------------

extern "C" void kernel_launch(void* const* d_in, const int* in_sizes, int n_in,
                              void* d_out, int out_size, void* d_ws, size_t ws_size,
                              hipStream_t stream) {
  (void)in_sizes; (void)n_in; (void)out_size; (void)ws_size;
  const float* x  = (const float*)d_in[0];
  const float* Wq = (const float*)d_in[1];
  const float* Wk = (const float*)d_in[2];
  const float* Wv = (const float*)d_in[3];
  const float* Wo = (const float*)d_in[4];
  const float* g  = (const float*)d_in[5];
  const float* bb = (const float*)d_in[6];

  float* out_y    = (float*)d_out;                       // [B,S,NIN]
  float* out_attn = out_y + (size_t)MTOT * NIN;          // [B,H,S,S]

  // workspace map (f16 buffers), ~88 MB total
  char* ws = (char*)d_ws;
  _Float16* xh   = (_Float16*)(ws + ((size_t)0  << 20));  // 16 MB [MTOT,NIN]
  _Float16* wq_t = (_Float16*)(ws + ((size_t)16 << 20));  //  2 MB [N,K]
  _Float16* wk_t = (_Float16*)(ws + ((size_t)18 << 20));
  _Float16* wv_t = (_Float16*)(ws + ((size_t)20 << 20));
  _Float16* wo_t = (_Float16*)(ws + ((size_t)22 << 20));
  _Float16* q    = (_Float16*)(ws + ((size_t)24 << 20));  // 16 MB [b,h,s,d]
  _Float16* k    = (_Float16*)(ws + ((size_t)40 << 20));  // 16 MB [b,h,s,d]
  _Float16* vt   = (_Float16*)(ws + ((size_t)56 << 20));  // 16 MB [b,h,d,s]
  _Float16* yh   = (_Float16*)(ws + ((size_t)72 << 20));  // 16 MB [MTOT,NIN]

  cvt_f32_to_f16<<<(MTOT * NIN) / 256, 256, 0, stream>>>(x, xh, MTOT * NIN);
  transpose_cvt<<<(NIN * NIN) / 256, 256, 0, stream>>>(Wq, wq_t);
  transpose_cvt<<<(NIN * NIN) / 256, 256, 0, stream>>>(Wk, wk_t);
  transpose_cvt<<<(NIN * NIN) / 256, 256, 0, stream>>>(Wv, wv_t);
  transpose_cvt<<<(NIN * NIN) / 256, 256, 0, stream>>>(Wo, wo_t);

  qkv_gemm<<<dim3(MTOT / 16, NIN / 64, 3), 128, 0, stream>>>(
      xh, wq_t, wk_t, wv_t, q, k, vt);

  attn_kernel<<<dim3(BATCH * NHEADC, SEQ / 16), 256, 0, stream>>>(
      q, k, vt, out_attn, yh);

  proj_ln<<<dim3(MTOT / 16), 256, 0, stream>>>(yh, wo_t, x, g, bb, out_y);
}
